// GatedRNN_24515673325782
// MI455X (gfx1250) — compile-verified, matrix operand order checked by reading, not run on hardware
//
#include <hip/hip_runtime.h>
#include <hip/hip_bf16.h>

// ---------------------------------------------------------------------------
// GatedRNN for MI455X (gfx1250, wave32, WMMA + async global->LDS).
// B=128,T=256,I=64,H=1024,O=64.
// 8 batch-groups (M=16) x 16 column-split WGs x 4 waves. Persistent kernel,
// 2 group barriers/step. h1/h2 exchanged via global bf16 mailboxes, staged
// into LDS with GLOBAL_LOAD_ASYNC_TO_LDS_B128 (ASYNCcnt) and consumed as
// conflict-free ds_load_b128 WMMA A-operands. Weights bf16-packed, L2-resident.
// ---------------------------------------------------------------------------

typedef unsigned short u16;
typedef __attribute__((ext_vector_type(16))) __bf16 v16bf;
typedef __attribute__((ext_vector_type(8)))  float  v8f;
typedef int v4i __attribute__((vector_size(16)));   // native b128 unit (AS-agnostic)

#define AS1 __attribute__((address_space(1)))
#define AS3 __attribute__((address_space(3)))

#define GROUPS  8
#define WGPG    16
#define TSTEPS  256
#define HDIM    1024
#define IDIM    64
#define ODIM    64
#define BROWS   16

constexpr int    LDSTRIDE   = HDIM + 8;                       // u16; 2064B row pitch -> bank-conflict-free b128
constexpr size_t SMEM_BYTES = (size_t)2 * BROWS * LDSTRIDE * 2; // h1 + h2 staged tiles = 66048B (< 320KB WGP LDS)

#if defined(__AMDGCN__) && __has_builtin(__builtin_amdgcn_global_load_async_to_lds_b128)
#define HAVE_ASYNC 1
#else
#define HAVE_ASYNC 0
#endif

// ---- workspace layout (bytes) ----
constexpr size_t SZ_PW_IH1 = (size_t)64 * 2  * 1024;
constexpr size_t SZ_PW_HH  = (size_t)64 * 32 * 1024;
constexpr size_t SZ_PW_G   = (size_t)64 * 64 * 1024;
constexpr size_t SZ_PW_O   = (size_t)4  * 32 * 1024;
constexpr size_t SZ_BIAS   = 4096 * sizeof(float);
constexpr size_t SZ_XB     = (size_t)128 * 256 * 64 * 2;
constexpr size_t SZ_HBUF   = (size_t)GROUPS * BROWS * HDIM * 2;  // single-buffered mailbox

constexpr size_t OFF_PW_IH1 = 0;
constexpr size_t OFF_PW_HH1 = OFF_PW_IH1 + SZ_PW_IH1;
constexpr size_t OFF_PW_IH2 = OFF_PW_HH1 + SZ_PW_HH;
constexpr size_t OFF_PW_HH2 = OFF_PW_IH2 + SZ_PW_HH;
constexpr size_t OFF_PW_G   = OFF_PW_HH2 + SZ_PW_HH;
constexpr size_t OFF_PW_O1  = OFF_PW_G   + SZ_PW_G;
constexpr size_t OFF_PW_O2  = OFF_PW_O1  + SZ_PW_O;
constexpr size_t OFF_BIAS   = OFF_PW_O2  + SZ_PW_O;
constexpr size_t OFF_XB     = OFF_BIAS   + SZ_BIAS;
constexpr size_t OFF_H1     = OFF_XB     + SZ_XB;
constexpr size_t OFF_H2     = OFF_H1     + SZ_HBUF;
constexpr size_t OFF_BAR    = OFF_H2     + SZ_HBUF;

// ---- helpers ----
__device__ __host__ inline u16 f2bf(float f) {
  union { float f; unsigned u; } x; x.f = f;
  unsigned r = x.u + 0x7FFFu + ((x.u >> 16) & 1u);   // RNE
  return (u16)(r >> 16);
}

__device__ inline float fast_tanh(float x) {
  float e = __builtin_amdgcn_exp2f(x * 2.885390082f);          // exp(2x)
  return 1.0f - 2.0f * __builtin_amdgcn_rcpf(e + 1.0f);
}
__device__ inline float fast_sigmoid(float x) {
  float e = __builtin_amdgcn_exp2f(-x * 1.442695041f);
  return __builtin_amdgcn_rcpf(1.0f + e);
}

// AS3 pointers are 32-bit; generic LDS addr[31:0] == LDS byte offset (ISA 10.2)
__device__ inline AS3 u16* to_lds(u16* p) {
  return (AS3 u16*)(unsigned)(unsigned long long)p;
}
__device__ inline AS1 v4i* glb_v4(const void* p) {
  return (AS1 v4i*)(unsigned long long)p;
}
__device__ inline AS3 v4i* lds_v4(AS3 u16* p) {
  return (AS3 v4i*)p;
}

union BfTile { v4i q[2]; v16bf v; };

// A operand from global row-major bf16 [16][ldA] (used for the small x tiles)
__device__ inline v16bf load_a_tile(const u16* __restrict__ A, int ldA, int kt, int lane) {
  int row   = lane & 15;
  int kbase = kt * 32 + (lane >> 4) * 8;
  const u16* p = A + row * ldA + kbase;
  BfTile t;
  t.q[0] = *(const v4i*)(p);          // K = kbase+0..7
  t.q[1] = *(const v4i*)(p + 16);     // K = kbase+16..23
  return t.v;
}

// B operand: pre-packed, 1KB per 32x16 K-tile, 32B per lane, contiguous.
__device__ inline v16bf load_b_tile(const u16* __restrict__ Bp, int tileIdx, int lane) {
  const u16* p = Bp + (size_t)tileIdx * 512 + lane * 16;
  BfTile t;
  t.q[0] = *(const v4i*)(p);
  t.q[1] = *(const v4i*)(p + 8);
  return t.v;
}

__device__ inline v8f vzero8() {
  v8f z;
#pragma unroll
  for (int i = 0; i < 8; ++i) z[i] = 0.0f;
  return z;
}

// acc += A(global)[16,nk*32] @ Bpacked
__device__ inline v8f gemm_acc(v8f acc, const u16* __restrict__ A, int ldA,
                               const u16* __restrict__ Bp, int tileBase, int nk, int lane) {
#pragma unroll 4
  for (int kt = 0; kt < nk; ++kt) {
    v16bf a = load_a_tile(A, ldA, kt, lane);
    v16bf b = load_b_tile(Bp, tileBase + kt, lane);
    acc = __builtin_amdgcn_wmma_f32_16x16x32_bf16(false, a, false, b,
                                                  (short)0, acc, false, false);
  }
  return acc;
}

// acc += A(LDS, padded pitch)[16,nk*32] @ Bpacked   -- ds_load_b128 A-operands
__device__ inline v8f gemm_acc_lds(v8f acc, AS3 const u16* A,
                                   const u16* __restrict__ Bp, int tileBase, int nk, int lane) {
  AS3 const u16* base = A + (lane & 15) * LDSTRIDE + (lane >> 4) * 8;
#pragma unroll 4
  for (int kt = 0; kt < nk; ++kt) {
    AS3 const u16* p = base + kt * 32;
    BfTile ta;
    ta.q[0] = *(AS3 const v4i*)(p);
    ta.q[1] = *(AS3 const v4i*)(p + 16);
    v16bf b = load_b_tile(Bp, tileBase + kt, lane);
    acc = __builtin_amdgcn_wmma_f32_16x16x32_bf16(false, ta.v, false, b,
                                                  (short)0, acc, false, false);
  }
  return acc;
}

// Cooperative 32KB stage: global h mailbox [16][1024] -> LDS [16][LDSTRIDE].
// 128 threads: thread -> (row = t/8, 128-col chunk = t%8), 16x b128 each.
__device__ inline void stage_h(AS3 u16* dst, const u16* __restrict__ src) {
  int row = threadIdx.x >> 3;
  int c0  = (threadIdx.x & 7) * 128;
  const u16* s = src + row * HDIM + c0;
  AS3 u16*   d = dst + row * LDSTRIDE + c0;
#pragma unroll
  for (int i = 0; i < 16; ++i) {
#if HAVE_ASYNC
    __builtin_amdgcn_global_load_async_to_lds_b128(glb_v4(s + i * 8),
                                                   lds_v4(d + i * 8), 0, 0);
#else
    *(AS3 v4i*)(d + i * 8) = *(const v4i*)(s + i * 8);
#endif
  }
}
__device__ inline void stage_wait() {
#if HAVE_ASYNC
#if __has_builtin(__builtin_amdgcn_s_wait_asynccnt)
  __builtin_amdgcn_s_wait_asynccnt(0);
#else
  asm volatile("s_wait_asynccnt 0x0" ::: "memory");
#endif
#endif
  __syncthreads();
}

__device__ inline void zero_lds_tile(AS3 u16* dst) {
  int row = threadIdx.x >> 3;
  int c0  = (threadIdx.x & 7) * 128;
  AS3 u16* d = dst + row * LDSTRIDE + c0;
  v4i z = {0, 0, 0, 0};
#pragma unroll
  for (int i = 0; i < 16; ++i) *(AS3 v4i*)(d + i * 8) = z;
}

// 16-workgroup group barrier via global atomics (L2 scope).
__device__ inline void group_barrier(unsigned* ctr, unsigned& target) {
  __threadfence();
  __syncthreads();
  target += WGPG;
  if (threadIdx.x == 0) {
    __hip_atomic_fetch_add(ctr, 1u, __ATOMIC_RELEASE, __HIP_MEMORY_SCOPE_AGENT);
    while (__hip_atomic_load(ctr, __ATOMIC_ACQUIRE, __HIP_MEMORY_SCOPE_AGENT) < target) {
      __builtin_amdgcn_s_sleep(2);
    }
  }
  __syncthreads();
  __threadfence();
}

// ---------------------------------------------------------------------------
// Pack kernels
// ---------------------------------------------------------------------------
__global__ void pack_w_kernel(const float* __restrict__ W, u16* __restrict__ dst,
                              int K, int kTiles, int nTiles) {
  int idx  = blockIdx.x * blockDim.x + threadIdx.x;
  int lane = idx & 31;
  int tile = idx >> 5;
  if (tile >= kTiles * nTiles) return;
  int kt = tile % kTiles, nt = tile / kTiles;
  int col   = nt * 16 + (lane & 15);
  int kbase = kt * 32 + (lane >> 4) * 8;
  u16* o = dst + (size_t)tile * 512 + lane * 16;
#pragma unroll
  for (int j = 0; j < 16; ++j) {
    int k = kbase + j + ((j >= 8) ? 8 : 0);
    o[j] = f2bf(W[(size_t)col * K + k]);      // B[k][n] = W[n][k]
  }
}

__global__ void pack_x_kernel(const float* __restrict__ x, u16* __restrict__ xb) {
  int idx = blockIdx.x * blockDim.x + threadIdx.x;
  if (idx >= 128 * 256 * 64) return;
  int i = idx % 64;
  int t = (idx / 64) % 256;
  int b = idx / (64 * 256);
  int g = b >> 4, r = b & 15;
  xb[(((size_t)g * 256 + t) * 16 + r) * 64 + i] = f2bf(x[idx]);
}

__global__ void pack_bias_kernel(const float* b_ih1, const float* b_hh1,
                                 const float* b_ih2, const float* b_hh2,
                                 const float* bg, const float* bo1, const float* bo2,
                                 float* dst, unsigned* bar) {
  int i = blockIdx.x * blockDim.x + threadIdx.x;
  if (i < HDIM) {
    dst[i]            = b_ih1[i] + b_hh1[i];
    dst[HDIM + i]     = b_ih2[i] + b_hh2[i];
    dst[2 * HDIM + i] = bg[i];
  }
  if (i < ODIM) {
    dst[3 * HDIM + i]        = bo1[i];
    dst[3 * HDIM + ODIM + i] = bo2[i];
  }
  if (i < GROUPS) bar[i] = 0u;
}

// ---------------------------------------------------------------------------
// Persistent RNN kernel: 128 blocks x 128 threads (4 waves)
// ---------------------------------------------------------------------------
__global__ __launch_bounds__(128, 1)
void rnn_kernel(const u16* __restrict__ pw_ih1, const u16* __restrict__ pw_hh1,
                const u16* __restrict__ pw_ih2, const u16* __restrict__ pw_hh2,
                const u16* __restrict__ pw_g,   const u16* __restrict__ pw_o1,
                const u16* __restrict__ pw_o2,  const float* __restrict__ biases,
                const u16* __restrict__ xb, u16* __restrict__ h1buf,
                u16* __restrict__ h2buf, unsigned* __restrict__ bar,
                float* __restrict__ out) {
  extern __shared__ __align__(16) u16 smem[];
  AS3 u16* lds_h1 = to_lds(smem);
  AS3 u16* lds_h2 = to_lds(smem + BROWS * LDSTRIDE);

  const int g    = blockIdx.x / WGPG;
  const int wg   = blockIdx.x % WGPG;
  const int wave = threadIdx.x >> 5;
  const int lane = threadIdx.x & 31;
  const int ct   = wg * 4 + wave;      // column tile 0..63 of H
  const int n0   = ct * 16;
  const int colL = lane & 15;
  const int rowHi = (lane >> 4) * 8;

  u16* h1x = h1buf + (size_t)g * BROWS * HDIM;   // single-buffer exchange mailbox
  u16* h2x = h2buf + (size_t)g * BROWS * HDIM;
  const u16* xg = xb + (size_t)g * TSTEPS * BROWS * IDIM;
  unsigned* ctr = bar + g;
  unsigned target = 0;

  const float b1l = biases[n0 + colL];
  const float b2l = biases[HDIM + n0 + colL];
  const float bgl = biases[2 * HDIM + n0 + colL];
  const bool  doOut = (wg == 0);
  const float bo1l = doOut ? biases[3 * HDIM + wave * 16 + colL] : 0.0f;
  const float bo2l = doOut ? biases[3 * HDIM + ODIM + wave * 16 + colL] : 0.0f;

  // initial h1 = h2 = 0 live in LDS only
  zero_lds_tile(lds_h1);
  zero_lds_tile(lds_h2);
  __syncthreads();

  v8f u, h2f;
#pragma unroll
  for (int r = 0; r < 8; ++r) { u[r] = 1.0f; h2f[r] = 0.0f; }

  for (int t = 0; t < TSTEPS; ++t) {
    if (t + 1 < TSTEPS)
      __builtin_prefetch(xg + (size_t)(t + 1) * BROWS * IDIM + threadIdx.x * 8, 0, 1);

    // ---- phase 1: h1 = tanh(x_t@W_ih1^T + h1_prev@W_hh1^T + b) ----
    v8f acc = vzero8();
    acc = gemm_acc(acc, xg + (size_t)t * BROWS * IDIM, IDIM, pw_ih1, ct * 2, 2, lane);
    acc = gemm_acc_lds(acc, lds_h1, pw_hh1, ct * 32, 32, lane);   // lds_h1 == h1[t-1]
#pragma unroll
    for (int r = 0; r < 8; ++r)
      h1x[(size_t)(r + rowHi) * HDIM + n0 + colL] = f2bf(fast_tanh(acc[r] + b1l));
    group_barrier(ctr, target);
    stage_h(lds_h1, h1x);            // async: full h1[t] -> LDS
    stage_wait();

    // ---- phase 2: h2 = u*tanh(h1@W_ih2^T + h2_prev@W_hh2^T + b) + (1-u)*h2 ----
    acc = vzero8();
    acc = gemm_acc_lds(acc, lds_h1, pw_ih2, ct * 32, 32, lane);
    acc = gemm_acc_lds(acc, lds_h2, pw_hh2, ct * 32, 32, lane);   // lds_h2 == h2[t-1]
#pragma unroll
    for (int r = 0; r < 8; ++r) {
      float hn  = fast_tanh(acc[r] + b2l);
      float h2v = u[r] * hn + (1.0f - u[r]) * h2f[r];
      h2f[r] = h2v;
      h2x[(size_t)(r + rowHi) * HDIM + n0 + colL] = f2bf(h2v);
    }
    group_barrier(ctr, target);
    stage_h(lds_h2, h2x);            // async: full h2[t] -> LDS
    stage_wait();

    // ---- phase 3: u_next = sigmoid([h1,h2]@Wg^T + bg); WG0 also emits out ----
    acc = vzero8();
    acc = gemm_acc_lds(acc, lds_h1, pw_g, ct * 64,      32, lane);
    acc = gemm_acc_lds(acc, lds_h2, pw_g, ct * 64 + 32, 32, lane);
#pragma unroll
    for (int r = 0; r < 8; ++r) u[r] = fast_sigmoid(acc[r] + bgl);

    if (doOut) {
      v8f a1 = vzero8(), a2 = vzero8();
      a1 = gemm_acc_lds(a1, lds_h1, pw_o1, wave * 32, 32, lane);
      a2 = gemm_acc_lds(a2, lds_h2, pw_o2, wave * 32, 32, lane);
      const int b0 = g * 16;
#pragma unroll
      for (int r = 0; r < 8; ++r) {
        float v = fast_tanh(a1[r] + bo1l) + fast_tanh(a2[r] + bo2l);
        out[(((size_t)(b0 + r + rowHi)) * TSTEPS + t) * ODIM + wave * 16 + colL] = v;
      }
    }
    // mailbox reuse is race-free: all stagers of h1/h2 pass the *next* group
    // barrier before any WG can overwrite the mailbox on the next step.
  }
}

// ---------------------------------------------------------------------------
extern "C" void kernel_launch(void* const* d_in, const int* in_sizes, int n_in,
                              void* d_out, int out_size, void* d_ws, size_t ws_size,
                              hipStream_t stream) {
  const float* x     = (const float*)d_in[0];
  const float* W_ih1 = (const float*)d_in[1];
  const float* b_ih1 = (const float*)d_in[2];
  const float* W_hh1 = (const float*)d_in[3];
  const float* b_hh1 = (const float*)d_in[4];
  const float* W_ih2 = (const float*)d_in[5];
  const float* b_ih2 = (const float*)d_in[6];
  const float* W_hh2 = (const float*)d_in[7];
  const float* b_hh2 = (const float*)d_in[8];
  const float* Wg    = (const float*)d_in[9];
  const float* bg    = (const float*)d_in[10];
  const float* Wo1   = (const float*)d_in[11];
  const float* bo1   = (const float*)d_in[12];
  const float* Wo2   = (const float*)d_in[13];
  const float* bo2   = (const float*)d_in[14];

  char* ws = (char*)d_ws;
  u16*   pw_ih1 = (u16*)(ws + OFF_PW_IH1);
  u16*   pw_hh1 = (u16*)(ws + OFF_PW_HH1);
  u16*   pw_ih2 = (u16*)(ws + OFF_PW_IH2);
  u16*   pw_hh2 = (u16*)(ws + OFF_PW_HH2);
  u16*   pw_g   = (u16*)(ws + OFF_PW_G);
  u16*   pw_o1  = (u16*)(ws + OFF_PW_O1);
  u16*   pw_o2  = (u16*)(ws + OFF_PW_O2);
  float* biases = (float*)(ws + OFF_BIAS);
  u16*   xb     = (u16*)(ws + OFF_XB);
  u16*   h1buf  = (u16*)(ws + OFF_H1);
  u16*   h2buf  = (u16*)(ws + OFF_H2);
  unsigned* bar = (unsigned*)(ws + OFF_BAR);

  auto blocksFor = [](int tiles) { return (tiles * 32 + 255) / 256; };

  pack_w_kernel<<<blocksFor(64 * 2),  256, 0, stream>>>(W_ih1, pw_ih1, IDIM,     2,  64);
  pack_w_kernel<<<blocksFor(64 * 32), 256, 0, stream>>>(W_hh1, pw_hh1, HDIM,     32, 64);
  pack_w_kernel<<<blocksFor(64 * 32), 256, 0, stream>>>(W_ih2, pw_ih2, HDIM,     32, 64);
  pack_w_kernel<<<blocksFor(64 * 32), 256, 0, stream>>>(W_hh2, pw_hh2, HDIM,     32, 64);
  pack_w_kernel<<<blocksFor(64 * 64), 256, 0, stream>>>(Wg,    pw_g,   2 * HDIM, 64, 64);
  pack_w_kernel<<<blocksFor(4 * 32),  256, 0, stream>>>(Wo1,   pw_o1,  HDIM,     32, 4);
  pack_w_kernel<<<blocksFor(4 * 32),  256, 0, stream>>>(Wo2,   pw_o2,  HDIM,     32, 4);
  pack_x_kernel<<<(128 * 256 * 64 + 255) / 256, 256, 0, stream>>>(x, xb);
  pack_bias_kernel<<<4, 256, 0, stream>>>(b_ih1, b_hh1, b_ih2, b_hh2, bg, bo1, bo2,
                                          biases, bar);

  rnn_kernel<<<GROUPS * WGPG, 128, SMEM_BYTES, stream>>>(
      pw_ih1, pw_hh1, pw_ih2, pw_hh2, pw_g, pw_o1, pw_o2,
      biases, xb, h1buf, h2buf, bar, (float*)d_out);
}
